// MyModel_87522843560395
// MI455X (gfx1250) — matrix operand order repaired and checked by reading, not compiled, not run on hardware
//
#include <hip/hip_runtime.h>

// ---------------------------------------------------------------------------
// out[b] = sum_k x[b,k] * W[k] + bias      x: 256 x 1375432 f32 (1.41 GB)
//
// Pure HBM-streaming problem (0.5 flop/byte): floor = 1.408GB / 23.3TB/s ~ 60us.
// Strategy: stream x with non-temporal b128 loads, accumulate through the
// CDNA5 matrix pipe (v_wmma_f32_16x16x32_bf16) so the VALU only does cvt +
// addressing.  W (5.5MB) stays RT-cached in the 192MB L2 (reused by 16 row
// tiles).  Deterministic fixed-order reductions (no float atomics).
// ---------------------------------------------------------------------------

typedef __attribute__((ext_vector_type(4)))  float  v4f;
typedef __attribute__((ext_vector_type(8)))  float  v8f;
typedef __attribute__((ext_vector_type(16))) __bf16 v16bf;

namespace {
constexpr int K_DIM         = 1375432;
constexpr int BATCH         = 256;
constexpr int ROWS_PER_TILE = 16;                     // one WMMA M-tile
constexpr int N_TILES       = BATCH / ROWS_PER_TILE;  // 16
constexpr int CHUNKS        = 256;                    // k-chunk blocks per row tile
constexpr int WAVES         = 8;                      // waves per block (256 thr)
constexpr int WAVES_PER_TILE = CHUNKS * WAVES;        // 2048
constexpr int KMAIN         = (K_DIM / 32) * 32;      // 1375424 (WMMA-covered)
constexpr int NGROUPS       = KMAIN / 32;             // 42982 32-wide k-groups
}

// Convert 4 f32 lanes into bf16 slots [o, o+4) of a v16bf operand.
__device__ __forceinline__ void pack4(v16bf& d, int o, v4f s) {
    d[o + 0] = (__bf16)s[0];
    d[o + 1] = (__bf16)s[1];
    d[o + 2] = (__bf16)s[2];
    d[o + 3] = (__bf16)s[3];
}

__global__ __launch_bounds__(256)
void onehot_gemv_wmma(const float* __restrict__ X,
                      const float* __restrict__ W,
                      float* __restrict__ partial)   // [BATCH][CHUNKS]
{
    const int lane = threadIdx.x & 31;
    const int wave = threadIdx.x >> 5;
    const int half = lane >> 4;      // A layout: half 0 -> K 0-7/16-23, half 1 -> K 8-15/24-31
    const int lrow = lane & 15;      // A layout: M = lrow
    const int cx   = blockIdx.x;     // k-chunk index
    const int ty   = blockIdx.y;     // row tile index
    const int row  = ty * ROWS_PER_TILE + lrow;

    const size_t xrow = (size_t)row * K_DIM;
    const int g = cx * WAVES + wave;          // wave id within row tile [0,2048)

    // B-operand mask: only column N==0 carries W; other 15 columns are zero,
    // so C columns 1..15 stay exactly 0 and column 0 holds the row dots.
    const float wm = (lrow == 0) ? 1.0f : 0.0f;

    v8f acc = {};                              // f32 C accumulator

    // Grid-stride over 32-wide k-groups: exact coverage of [0, KMAIN),
    // uniform per-wave trip count -> EXEC is all-1s at every WMMA.
    for (int grp = g; grp < NGROUPS; grp += WAVES_PER_TILE) {
        const int kb = grp * 32;

        // ---- A: 16 rows x 32 k of x, this lane's 16 f32 (4 x b128, NT) ----
        const float* xp = X + xrow + kb + half * 8;
        v4f x0 = __builtin_nontemporal_load((const v4f*)(xp));
        v4f x1 = __builtin_nontemporal_load((const v4f*)(xp + 4));
        v4f x2 = __builtin_nontemporal_load((const v4f*)(xp + 16));
        v4f x3 = __builtin_nontemporal_load((const v4f*)(xp + 20));

        // ---- B: W[kb .. kb+31] broadcast per half, masked to column 0 ----
        const float* wp = W + kb + half * 8;  // lanes of a half share addrs (L2/RT)
        v4f w0 = *(const v4f*)(wp +  0) * wm;
        v4f w1 = *(const v4f*)(wp +  4) * wm;
        v4f w2 = *(const v4f*)(wp + 16) * wm;
        v4f w3 = *(const v4f*)(wp + 20) * wm;

        v16bf a, b;
        pack4(a, 0, x0);  pack4(a, 4, x1);  pack4(a,  8, x2);  pack4(a, 12, x3);
        pack4(b, 0, w0);  pack4(b, 4, w1);  pack4(b,  8, w2);  pack4(b, 12, w3);

        acc = __builtin_amdgcn_wmma_f32_16x16x32_bf16(
                  /*neg_a=*/false, a, /*neg_b=*/false, b,
                  /*c_mod=*/(short)0, acc, /*reuse_a=*/false, /*reuse_b=*/false);
    }

    // C column 0 lives in lane 0 (M = 0..7 in VGPR 0..7) and lane 16 (M = 8..15).
    __shared__ float part[WAVES][ROWS_PER_TILE];
    if (lane == 0) {
        #pragma unroll
        for (int r = 0; r < 8; ++r) part[wave][r] = acc[r];
    } else if (lane == 16) {
        #pragma unroll
        for (int r = 0; r < 8; ++r) part[wave][8 + r] = acc[r];
    }
    __syncthreads();

    // Fixed-order cross-wave reduction -> one partial per (row, chunk).
    if (threadIdx.x < ROWS_PER_TILE) {
        float s = 0.0f;
        #pragma unroll
        for (int w = 0; w < WAVES; ++w) s += part[w][threadIdx.x];
        partial[(size_t)(ty * ROWS_PER_TILE + threadIdx.x) * CHUNKS + cx] = s;
    }
}

__global__ __launch_bounds__(256)
void onehot_finalize(const float* __restrict__ partial,
                     const float* __restrict__ X,
                     const float* __restrict__ W,
                     const float* __restrict__ bias,
                     float* __restrict__ out)
{
    const int row = threadIdx.x;               // single block, 256 threads
    float s = 0.0f;
    const float* p = partial + (size_t)row * CHUNKS;
    #pragma unroll 4
    for (int c = 0; c < CHUNKS; c += 4) {
        v4f v = *(const v4f*)(p + c);
        s += v[0] + v[1] + v[2] + v[3];
    }
    // K remainder not covered by the 32-wide WMMA groups (8 elements).
    const size_t xrow = (size_t)row * K_DIM;
    #pragma unroll
    for (int k = KMAIN; k < K_DIM; ++k) s += X[xrow + k] * W[k];
    out[row] = s + bias[0];
}

extern "C" void kernel_launch(void* const* d_in, const int* in_sizes, int n_in,
                              void* d_out, int out_size, void* d_ws, size_t ws_size,
                              hipStream_t stream)
{
    (void)in_sizes; (void)n_in; (void)out_size; (void)ws_size;
    const float* X    = (const float*)d_in[0];
    const float* W    = (const float*)d_in[1];
    const float* bias = (const float*)d_in[2];
    float* partial    = (float*)d_ws;          // BATCH*CHUNKS*4 = 256 KB scratch
    float* out        = (float*)d_out;

    dim3 grid(CHUNKS, N_TILES);                // 4096 blocks x 8 waves = 32768 waves
    onehot_gemv_wmma<<<grid, 256, 0, stream>>>(X, W, partial);
    onehot_finalize<<<1, 256, 0, stream>>>(partial, X, W, bias, out);
}
